// Upsampling_17867063951707
// MI455X (gfx1250) — matrix operand order repaired and checked by reading, not compiled, not run on hardware
//
#include <hip/hip_runtime.h>
#include <hip/hip_bf16.h>
#include <math.h>
#include <stdint.h>

typedef __attribute__((ext_vector_type(16))) __bf16 v16bf;
typedef __attribute__((ext_vector_type(8)))  float  v8f;
typedef __attribute__((ext_vector_type(4)))  unsigned int v4u;
typedef __attribute__((ext_vector_type(8)))  int    v8i;
typedef __attribute__((ext_vector_type(4)))  int    v4i;

#define CIN  64
#define COUT 64
#define KS   5
#define SDIM 3

// ---------------- zero A + deg ----------------
__global__ void k_zero(float* __restrict__ A, float* __restrict__ deg, size_t nA, int Nx) {
  size_t i = (size_t)blockIdx.x * blockDim.x + threadIdx.x;
  size_t stride = (size_t)gridDim.x * blockDim.x;
  size_t n4 = nA >> 2;
  for (size_t j = i; j < n4; j += stride) ((float4*)A)[j] = make_float4(0.f, 0.f, 0.f, 0.f);
  for (size_t j = i; j < (size_t)Nx; j += stride) deg[j] = 0.f;
}

// ---------------- edge scatter: A[dst, kidx, c] += basis * x[src, c] ----------------
__global__ void k_scatter(const float* __restrict__ x, const int* __restrict__ ei,
                          const float* __restrict__ ea, float* __restrict__ A,
                          float* __restrict__ deg, int E, int KALL) {
  int e = blockIdx.x * 4 + (threadIdx.x >> 6);   // 4 edges per 256-thread block
  int c = threadIdx.x & 63;                      // channel
  if (e >= E) return;
  int src = ei[e];
  int dst = ei[E + e];
  float xv = x[(size_t)src * CIN + c];
  float fr[SDIM]; int lo[SDIM];
#pragma unroll
  for (int d = 0; d < SDIM; ++d) {
    float v = ea[(size_t)e * SDIM + d] * (float)(KS - 1);
    float fl = floorf(v);
    lo[d] = (int)fl;
    fr[d] = v - fl;
  }
  if (c == 0) atomicAdd(&deg[dst], 1.0f);
  float* Arow = A + (size_t)dst * KALL;
#pragma unroll
  for (int s = 0; s < 8; ++s) {
    float basis = 1.f; int kidx = 0; int mult = 1;
#pragma unroll
    for (int d = 0; d < SDIM; ++d) {
      int b = (s >> d) & 1;
      basis *= b ? fr[d] : (1.f - fr[d]);
      int id = lo[d] + b;
      id = id < 0 ? 0 : (id > KS - 1 ? KS - 1 : id);
      kidx += id * mult;
      mult *= KS;
    }
    atomicAdd(&Arow[kidx * CIN + c], basis * xv);
  }
}

// ---------------- append x * max(deg,1) as extra K columns ----------------
__global__ void k_xcols(const float* __restrict__ x, const float* __restrict__ deg,
                        float* __restrict__ A, int Nx, int KALL, int Ktot) {
  int i = blockIdx.x * blockDim.x + threadIdx.x;
  if (i >= Nx * CIN) return;
  int n = i / CIN, ci = i % CIN;
  float dg = fmaxf(deg[n], 1.f);
  A[(size_t)n * KALL + (size_t)Ktot * CIN + ci] = x[i] * dg;
}

// ------- pack [weight ; root] into per-lane WMMA B-fragment order (bf16) -------
// Bpack[((kb*4+nt)*32+lane)*16 + e] = B[kb*32 + (lane<16?0:16) + e][nt*16 + lane%16]
__global__ void k_pack(const float* __restrict__ w, const float* __restrict__ root,
                       __bf16* __restrict__ bp, int KB, int Kw) {
  int t = blockIdx.x * blockDim.x + threadIdx.x;  // (kb*4+nt)*32 + lane
  if (t >= KB * 128) return;
  int lane = t & 31;
  int nt   = (t >> 5) & 3;
  int kb   = t >> 7;
  int n = nt * 16 + (lane & 15);
  int kbase = kb * 32 + ((lane >> 4) << 4);
  __bf16* o = bp + (size_t)t * 16;
#pragma unroll
  for (int e2 = 0; e2 < 16; ++e2) {
    int K = kbase + e2;
    float val = (K < Kw) ? w[(size_t)K * COUT + n] : root[(size_t)(K - Kw) * COUT + n];
    o[e2] = (__bf16)val;
  }
}

// ---------------- TDM: 2D tile (32 K-cols x 64 rows of f32) -> LDS ----------------
__device__ __forceinline__ void tdm_load_tile(unsigned lds_addr,
                                              unsigned long long gaddr,
                                              unsigned stride_elems) {
  v4u g0;
  g0[0] = 1u;                                             // count=1, user descriptor
  g0[1] = lds_addr;                                       // LDS byte address
  g0[2] = (unsigned)(gaddr & 0xffffffffull);              // global_addr[31:0]
  g0[3] = (unsigned)((gaddr >> 32) & 0x01ffffffull)       // global_addr[56:32]
          | (2u << 30);                                   // type = 2 ("image")
  const unsigned td0 = 0x40000000u, td1 = 0x40000000u;    // tensor dims: tile always in-bounds
  const unsigned tile0 = 32u, tile1 = 64u, tile2 = 0u;
  unsigned long long s0 = (unsigned long long)stride_elems;  // dim0 stride (data_size units)
  v8i g1;
  g1[0] = (int)(2u << 16);                                // data_size=2 -> 4 bytes
  g1[1] = (int)((td0 & 0xffffu) << 16);                   // tensor_dim0[15:0]
  g1[2] = (int)(((td0 >> 16) & 0xffffu) | ((td1 & 0xffffu) << 16));
  g1[3] = (int)(((td1 >> 16) & 0xffffu) | ((tile0 & 0xffffu) << 16));
  g1[4] = (int)((tile1 & 0xffffu) | ((tile2 & 0xffffu) << 16));
  g1[5] = (int)(s0 & 0xffffffffull);                      // tensor_dim0_stride[31:0]
  g1[6] = (int)((s0 >> 32) & 0xffffull);                  // stride[47:32]; dim1_stride[15:0]=0
  g1[7] = 0;
  v4i gz = {0, 0, 0, 0};
#if defined(__clang_major__) && (__clang_major__ >= 23)
  v8i gz8 = {0, 0, 0, 0, 0, 0, 0, 0};
  __builtin_amdgcn_tensor_load_to_lds(g0, g1, gz, gz, gz8, 0);
#else
  __builtin_amdgcn_tensor_load_to_lds(g0, g1, gz, gz, 0);
#endif
}

// ------- WMMA GEMM (TDM -> LDS double buffer -> bf16 WMMA):
//         h = elu( (A @ [W;root]) / max(deg,1) + bias ) -------
__global__ void __launch_bounds__(256)
k_gemm(const float* __restrict__ A, const __bf16* __restrict__ bp,
       const float* __restrict__ deg, const float* __restrict__ bias,
       float* __restrict__ h, int KALL) {
  __shared__ float sA[2][64 * 32];                 // 16 KB double buffer
  int lane = threadIdx.x & 31;
  int wave = threadIdx.x >> 5;
  int mi = wave >> 1;                              // 0..3 : M tile within block
  int nh = wave & 1;                               // 0..1 : N half (32 cols)
  int rowBlock = blockIdx.x * 64;
  int KB = KALL / 32;

  const float* sBase0 = &sA[0][(mi * 16 + (lane & 15)) * 32 + ((lane >> 4) << 4)];
  const float* sBase1 = sBase0 + 64 * 32;
  const __bf16* bcol0 = bp + ((size_t)(nh * 2) * 32 + lane) * 16;   // nt = 2*nh
  const __bf16* bcol1 = bcol0 + 32 * 16;                            // nt = 2*nh+1

  unsigned lds0 = (unsigned)(uintptr_t)(&sA[0][0]);
  unsigned lds1 = (unsigned)(uintptr_t)(&sA[1][0]);
  unsigned long long gbase =
      (unsigned long long)(uintptr_t)(A + (size_t)rowBlock * KALL);

  v8f acc0 = {}, acc1 = {};
  if (wave == 0) {
    tdm_load_tile(lds0, gbase, (unsigned)KALL);
    __builtin_amdgcn_s_wait_tensorcnt(0);
  }
  __syncthreads();

  for (int kb = 0; kb < KB; ++kb) {
    int cur = kb & 1;
    if (wave == 0 && (kb + 1) < KB)                 // prefetch next tile into other buffer
      tdm_load_tile(cur ? lds0 : lds1,
                    gbase + (unsigned long long)(kb + 1) * 32ull * 4ull,
                    (unsigned)KALL);
    const float* sp = cur ? sBase1 : sBase0;
    float4 f0 = *(const float4*)(sp);
    float4 f1 = *(const float4*)(sp + 4);
    float4 f2 = *(const float4*)(sp + 8);
    float4 f3 = *(const float4*)(sp + 12);
    v16bf a;
    a[0]=(__bf16)f0.x; a[1]=(__bf16)f0.y; a[2]=(__bf16)f0.z; a[3]=(__bf16)f0.w;
    a[4]=(__bf16)f1.x; a[5]=(__bf16)f1.y; a[6]=(__bf16)f1.z; a[7]=(__bf16)f1.w;
    a[8]=(__bf16)f2.x; a[9]=(__bf16)f2.y; a[10]=(__bf16)f2.z; a[11]=(__bf16)f2.w;
    a[12]=(__bf16)f3.x; a[13]=(__bf16)f3.y; a[14]=(__bf16)f3.z; a[15]=(__bf16)f3.w;
    __builtin_prefetch(bcol0 + (size_t)(kb + 1) * 2048, 0, 1);      // global_prefetch_b8
    v16bf b0 = *(const v16bf*)(bcol0 + (size_t)kb * 2048);
    v16bf b1 = *(const v16bf*)(bcol1 + (size_t)kb * 2048);
    acc0 = __builtin_amdgcn_wmma_f32_16x16x32_bf16(false, a, false, b0,
                                                   (short)0, acc0, false, false);
    acc1 = __builtin_amdgcn_wmma_f32_16x16x32_bf16(false, a, false, b1,
                                                   (short)0, acc1, false, false);
    __syncthreads();                                // everyone done reading buf[cur]
    if (wave == 0) __builtin_amdgcn_s_wait_tensorcnt(0);
    __syncthreads();                                // next buffer published
  }

#pragma unroll
  for (int r = 0; r < 8; ++r) {
    int m = rowBlock + mi * 16 + ((lane >> 4) << 3) + r;
    float dg = fmaxf(deg[m], 1.f);
    int n0 = nh * 32 + (lane & 15);
    float v0 = acc0[r] / dg + bias[n0];
    v0 = v0 > 0.f ? v0 : expm1f(v0);                // ELU (alpha=1)
    h[(size_t)m * COUT + n0] = v0;
    int n1 = n0 + 16;
    float v1 = acc1[r] / dg + bias[n1];
    v1 = v1 > 0.f ? v1 : expm1f(v1);
    h[(size_t)m * COUT + n1] = v1;
  }
}

// ---------------- kNN (k<=8) + inverse-d2 interpolation ----------------
__global__ void k_knn(const float* __restrict__ h, const float* __restrict__ px,
                      const float* __restrict__ py, const int* __restrict__ bx,
                      const int* __restrict__ by, const int* __restrict__ kptr,
                      float* __restrict__ out, int Nx, int Ny) {
  int q = blockIdx.x * blockDim.x + threadIdx.x;
  if (q >= Ny) return;
  int kk = kptr[0]; kk = kk < 1 ? 1 : (kk > 8 ? 8 : kk);
  float y0 = py[(size_t)q*3+0], y1 = py[(size_t)q*3+1], y2 = py[(size_t)q*3+2];
  float yy = y0*y0 + y1*y1 + y2*y2;
  int myb = by[q];
  float bd[8]; int bi[8];
#pragma unroll
  for (int i = 0; i < 8; ++i) { bd[i] = 3.0e38f; bi[i] = 0; }
  for (int j = 0; j < Nx; ++j) {
    float x0 = px[(size_t)j*3+0], x1 = px[(size_t)j*3+1], x2 = px[(size_t)j*3+2];
    float d2 = yy + (x0*x0 + x1*x1 + x2*x2) - 2.0f*(y0*x0 + y1*x1 + y2*x2);
    if (bx[j] != myb) d2 = 1.0e10f;
    if (d2 < bd[kk-1]) {
      int p = kk - 1;
      while (p > 0 && bd[p-1] > d2) { bd[p] = bd[p-1]; bi[p] = bi[p-1]; --p; }
      bd[p] = d2; bi[p] = j;
    }
  }
  float w[8]; float ws = 0.f;
  for (int i = 0; i < kk; ++i) { w[i] = 1.0f / fmaxf(bd[i], 1.0e-16f); ws += w[i]; }
  float inv = 1.0f / ws;
  for (int c = 0; c < COUT; ++c) {
    float num = 0.f;
    for (int i = 0; i < kk; ++i) num += w[i] * h[(size_t)bi[i] * COUT + c];
    out[(size_t)q * COUT + c] = num * inv;
  }
}

extern "C" void kernel_launch(void* const* d_in, const int* in_sizes, int n_in,
                              void* d_out, int out_size, void* d_ws, size_t ws_size,
                              hipStream_t stream) {
  const float* x    = (const float*)d_in[0];
  const float* px   = (const float*)d_in[1];
  const float* py   = (const float*)d_in[2];
  const int*   ei   = (const int*)d_in[3];
  const float* ea   = (const float*)d_in[4];
  const int*   bx   = (const int*)d_in[5];
  const int*   by   = (const int*)d_in[6];
  const float* wgt  = (const float*)d_in[7];
  const float* root = (const float*)d_in[8];
  const float* bias = (const float*)d_in[9];
  const int*   kptr = (const int*)d_in[10];
  float* out = (float*)d_out;

  int Nx   = in_sizes[0] / CIN;            // 4096
  int Ny   = in_sizes[2] / SDIM;           // 16384
  int E    = in_sizes[3] / 2;              // 65536
  int Ktot = in_sizes[7] / (CIN * COUT);   // 125
  int KALL = (Ktot + 1) * CIN;             // 8064 (spline K + root K)
  int KB   = KALL / 32;                    // 252
  size_t nA = (size_t)Nx * KALL;

  auto align256 = [](size_t v) { return (v + 255) & ~(size_t)255; };
  char* ws = (char*)d_ws;
  float*  A   = (float*)ws;
  size_t  off = align256(nA * 4);
  float*  deg = (float*)(ws + off);  off = align256(off + (size_t)Nx * 4);
  __bf16* bp  = (__bf16*)(ws + off); off = align256(off + (size_t)KALL * COUT * 2);
  float*  h   = (float*)(ws + off);
  (void)ws_size; (void)n_in; (void)out_size;

  k_zero   <<<2048, 256, 0, stream>>>(A, deg, nA, Nx);
  k_scatter<<<(E + 3) / 4, 256, 0, stream>>>(x, ei, ea, A, deg, E, KALL);
  k_xcols  <<<(Nx * CIN + 255) / 256, 256, 0, stream>>>(x, deg, A, Nx, KALL, Ktot);
  k_pack   <<<(KB * 128 + 255) / 256, 256, 0, stream>>>(wgt, root, bp, KB, Ktot * CIN);
  k_gemm   <<<Nx / 64, 256, 0, stream>>>(A, bp, deg, bias, h, KALL);
  k_knn    <<<(Ny + 127) / 128, 128, 0, stream>>>(h, px, py, bx, by, kptr, out, Nx, Ny);
}
